// FunctionalGraphNetwork_46325517254763
// MI455X (gfx1250) — matrix operand order repaired and checked by reading, not compiled, not run on hardware
//
#include <hip/hip_runtime.h>
#include <hip/hip_bf16.h>

typedef __attribute__((ext_vector_type(16))) __bf16 v16bf;
typedef __attribute__((ext_vector_type(8)))  __bf16 v8bf;
typedef __attribute__((ext_vector_type(8)))  float  v8f;

#define NN    1024
#define LT    60                 // LSQ*T = 4*15
#define KTOT  (NN * LT)          // 61440
#define KG    16                 // K-groups in stage 2
#define KSLICE (KTOT / KG)       // 3840 K per group
#define PH_STEPS 4               // k-steps (of 32) per LDS phase
#define NPH   (KSLICE / (32 * PH_STEPS))   // 30 phases
#define LEAK  0.2f

// ---- CDNA5 async copy: global -> LDS (16B), tracked with ASYNCcnt --------------
__device__ __forceinline__ void async_ld16(unsigned lds_off, const char* gaddr) {
    asm volatile("global_load_async_to_lds_b128 %0, %1, off"
                 :: "v"(lds_off), "v"(gaddr) : "memory");
}
__device__ __forceinline__ void wait_async0() {
    asm volatile("s_wait_asynccnt 0x0" ::: "memory");
}

// ---------------- Stage 0a: x (f32) -> xbf (bf16, same row-major layout) --------
__global__ void __launch_bounds__(256) k_cvt_x(const float* __restrict__ x,
                                               __bf16* __restrict__ xbf) {
    int tid = blockIdx.x * 256 + threadIdx.x;          // 65536
    xbf[tid] = (__bf16)x[tid];
}

// ---------------- Stage 0b: W[i][j][lt] (f32) -> wpk bf16 in WMMA-B layout ------
// wpk[lt*4096 + ((i>>4)*64 + j)*16 + (i&15)]
__global__ void __launch_bounds__(256) k_pack_w(const float* __restrict__ W,
                                                __bf16* __restrict__ wpk) {
    int tid = blockIdx.x * 256 + threadIdx.x;          // 245760
    int lt = tid % LT;
    int ij = tid / LT;
    int j  = ij & 63;
    int i  = ij >> 6;
    wpk[lt * 4096 + ((i >> 4) * 64 + j) * 16 + (i & 15)] = (__bf16)W[tid];
}

// ---------------- Stage 1 (WMMA): y[k=(n,lt)][j] = sum_i x[n,i]*W[i,j,lt] -------
// y packed for stage-2 B: y[(k>>4)*1024 + j*16 + (k&15)]  (bf16)
// wave-job = (mtile 0..63, lt 0..59): 16 rows of n, K = 64 (2 wmma steps), 4 j-tiles
__global__ void __launch_bounds__(256) k_stage1(const __bf16* __restrict__ xbf,
                                                const __bf16* __restrict__ wpk,
                                                __bf16* __restrict__ y) {
    const int tid  = threadIdx.x;
    const int wave = tid >> 5;
    const int lane = tid & 31;
    const int laneM = lane & 15;
    const int hi    = lane >> 4;
    const int job   = blockIdx.x * 8 + wave;           // 0..3839
    const int mtile = job / LT;
    const int lt    = job % LT;
    const int n0    = mtile * 16;

    const __bf16* xrow = xbf + (size_t)(n0 + laneM) * 64;
    const __bf16* wlt  = wpk + (size_t)lt * 4096;

    v8f acc0 = {}, acc1 = {}, acc2 = {}, acc3 = {};
#pragma unroll
    for (int ks = 0; ks < 64; ks += 32) {
        // A fragment: lane=M(n); elems0-15: K=ks+hi*16+e
        v8bf alo = *(const v8bf*)(xrow + ks + hi * 16);
        v8bf ahi = *(const v8bf*)(xrow + ks + hi * 16 + 8);
        v16bf a = __builtin_shufflevector(alo, ahi, 0,1,2,3,4,5,6,7,8,9,10,11,12,13,14,15);
        // B fragments: lane=N(j); K = ks + hi*16 + e  -> kblk = (ks>>4)+hi
        const __bf16* bb = wlt + (((ks >> 4) + hi) * 64 + laneM) * 16;
        v16bf b0 = *(const v16bf*)(bb + 0 * 256);
        v16bf b1 = *(const v16bf*)(bb + 1 * 256);
        v16bf b2 = *(const v16bf*)(bb + 2 * 256);
        v16bf b3 = *(const v16bf*)(bb + 3 * 256);
        acc0 = __builtin_amdgcn_wmma_f32_16x16x32_bf16(false, a, false, b0, (short)0, acc0, false, false);
        acc1 = __builtin_amdgcn_wmma_f32_16x16x32_bf16(false, a, false, b1, (short)0, acc1, false, false);
        acc2 = __builtin_amdgcn_wmma_f32_16x16x32_bf16(false, a, false, b2, (short)0, acc2, false, false);
        acc3 = __builtin_amdgcn_wmma_f32_16x16x32_bf16(false, a, false, b3, (short)0, acc3, false, false);
    }
    // scatter into packed-B layout: C/D elem e -> n = n0 + e + 8*hi, j = jt*16+laneM
#pragma unroll
    for (int e = 0; e < 8; ++e) {
        int n = n0 + e + 8 * hi;
        int k = n * LT + lt;
        __bf16* yb = y + (size_t)(k >> 4) * 1024 + (k & 15);
        yb[(0 * 16 + laneM) * 16] = (__bf16)acc0[e];
        yb[(1 * 16 + laneM) * 16] = (__bf16)acc1[e];
        yb[(2 * 16 + laneM) * 16] = (__bf16)acc2[e];
        yb[(3 * 16 + laneM) * 16] = (__bf16)acc3[e];
    }
}

// ---------------- Stage 2: part[kg][m][j] = sum_{k in slice} adj[m,k]*y[k,j] ----
// block = (mgroup 0..7, kg 0..15); 8 waves, wave w owns m-tile mgroup*8+w.
// All waves share one K-slice -> B tiles staged in LDS via async loads.
__global__ void __launch_bounds__(256) k_stage2(const float* __restrict__ adj,
                                                const __bf16* __restrict__ y,
                                                float* __restrict__ part) {
    __shared__ __align__(128) char bstage[2][PH_STEPS * 4096];   // 2 x 16KB
    const int mgroup = blockIdx.x;                      // 0..7
    const int kg     = blockIdx.y;                      // 0..KG-1
    const int tid    = threadIdx.x;
    const int wave   = tid >> 5;
    const int lane   = tid & 31;
    const int laneM  = lane & 15;
    const int hi     = lane >> 4;
    const int m0w    = (mgroup * 8 + wave) * 16;

    const size_t kbase = (size_t)kg * KSLICE;
    const float* arow  = adj + (size_t)(m0w + laneM) * KTOT + kbase + hi * 16;
    const char*  ysrc  = (const char*)y + kbase * 128;  // bytes: (kbase/16)*1024*2

    const unsigned lds0 = (unsigned)(unsigned long long)(void*)&bstage[0][0];
    const unsigned ldsT = tid * 64;                      // 256 thr * 64B = 16KB/phase

    // preload phase 0
    {
        const char* g = ysrc + ldsT;
#pragma unroll
        for (int c = 0; c < 4; ++c) async_ld16(lds0 + ldsT + c * 16, g + c * 16);
    }
    wait_async0();
    __syncthreads();

    v8f acc0 = {}, acc1 = {}, acc2 = {}, acc3 = {};

    for (int p = 0; p < NPH; ++p) {
        if (p + 1 < NPH) {                               // prefetch next phase
            const char* g = ysrc + (size_t)(p + 1) * (PH_STEPS * 4096) + ldsT;
            unsigned l = lds0 + ((p + 1) & 1) * (PH_STEPS * 4096) + ldsT;
#pragma unroll
            for (int c = 0; c < 4; ++c) async_ld16(l + c * 16, g + c * 16);
        }
        const char*  tile = &bstage[p & 1][0];
        const float* aph  = arow + p * (PH_STEPS * 32);
#pragma unroll
        for (int s = 0; s < PH_STEPS; ++s) {
            // A fragment from global (f32 -> bf16 in regs)
            const float4* ap = (const float4*)(aph + s * 32);
            float4 f0 = ap[0], f1 = ap[1], f2 = ap[2], f3 = ap[3];
            v16bf a;
            a[0]  = (__bf16)f0.x; a[1]  = (__bf16)f0.y; a[2]  = (__bf16)f0.z; a[3]  = (__bf16)f0.w;
            a[4]  = (__bf16)f1.x; a[5]  = (__bf16)f1.y; a[6]  = (__bf16)f1.z; a[7]  = (__bf16)f1.w;
            a[8]  = (__bf16)f2.x; a[9]  = (__bf16)f2.y; a[10] = (__bf16)f2.z; a[11] = (__bf16)f2.w;
            a[12] = (__bf16)f3.x; a[13] = (__bf16)f3.y; a[14] = (__bf16)f3.z; a[15] = (__bf16)f3.w;
            // B fragments from LDS: offset = (s*2+hi)*2048 + jt*512 + laneM*32
            const char* tb = tile + s * 4096 + hi * 2048 + laneM * 32;
            v16bf b0 = *(const v16bf*)(tb + 0 * 512);
            v16bf b1 = *(const v16bf*)(tb + 1 * 512);
            v16bf b2 = *(const v16bf*)(tb + 2 * 512);
            v16bf b3 = *(const v16bf*)(tb + 3 * 512);
            acc0 = __builtin_amdgcn_wmma_f32_16x16x32_bf16(false, a, false, b0, (short)0, acc0, false, false);
            acc1 = __builtin_amdgcn_wmma_f32_16x16x32_bf16(false, a, false, b1, (short)0, acc1, false, false);
            acc2 = __builtin_amdgcn_wmma_f32_16x16x32_bf16(false, a, false, b2, (short)0, acc2, false, false);
            acc3 = __builtin_amdgcn_wmma_f32_16x16x32_bf16(false, a, false, b3, (short)0, acc3, false, false);
        }
        wait_async0();          // my prefetches landed
        __syncthreads();        // everyone done reading cur + writing nxt
    }

    // store this wave's m-tile: elem e -> m = m0w + e + 8*hi, j = jt*16 + laneM
    float* pbase = part + ((size_t)kg * 1024 + m0w) * 64;
#pragma unroll
    for (int e = 0; e < 8; ++e) {
        int ml = e + 8 * hi;
        pbase[ml * 64 +  0 + laneM] = acc0[e];
        pbase[ml * 64 + 16 + laneM] = acc1[e];
        pbase[ml * 64 + 32 + laneM] = acc2[e];
        pbase[ml * 64 + 48 + laneM] = acc3[e];
    }
}

// ---------------- Stage 3: out = lrelu(sum_g part[g] + x @ W2) ------------------
__global__ void __launch_bounds__(256) k_epilogue(const float* __restrict__ part,
                                                  const float* __restrict__ x,
                                                  const float* __restrict__ W2,
                                                  float* __restrict__ out) {
    int idx = blockIdx.x * 256 + threadIdx.x;          // 65536 = 1024*64
    int m = idx >> 6;
    int j = idx & 63;
    float v = 0.f;
#pragma unroll
    for (int g = 0; g < KG; ++g) v += part[(size_t)g * 65536 + idx];
    const float* xr = x + m * 64;
#pragma unroll
    for (int i = 0; i < 64; ++i) v += xr[i] * W2[i * 64 + j];
    out[idx] = fmaxf(v, LEAK * v);
}

extern "C" void kernel_launch(void* const* d_in, const int* in_sizes, int n_in,
                              void* d_out, int out_size, void* d_ws, size_t ws_size,
                              hipStream_t stream) {
    (void)in_sizes; (void)n_in; (void)out_size; (void)ws_size;
    const float* x   = (const float*)d_in[0];   // (1024, 64)
    const float* adj = (const float*)d_in[1];   // (1024, 1024, 4, 15)
    const float* W   = (const float*)d_in[2];   // (64, 64, 4, 15)
    const float* W2  = (const float*)d_in[3];   // (64, 64)
    float* out = (float*)d_out;

    char* ws = (char*)d_ws;
    __bf16* xbf  = (__bf16*)ws;                                  //   131,072 B
    __bf16* wpk  = (__bf16*)(ws + 131072);                       //   491,520 B
    __bf16* y    = (__bf16*)(ws + 622592);                       // 7,864,320 B
    float*  part = (float*)(ws + 8486912);                       // 4,194,304 B

    k_cvt_x   <<<256, 256, 0, stream>>>(x, xbf);
    k_pack_w  <<<960, 256, 0, stream>>>(W, wpk);
    k_stage1  <<<480, 256, 0, stream>>>(xbf, wpk, y);
    k_stage2  <<<dim3(8, KG), 256, 0, stream>>>(adj, y, part);
    k_epilogue<<<256, 256, 0, stream>>>(part, x, W2, out);
}
